// HyperbolicLoRA_44470091382812
// MI455X (gfx1250) — matrix operand order repaired and checked by reading, not compiled, not run on hardware
//
#include <hip/hip_runtime.h>
#include <math.h>

// ---------------------------------------------------------------------------
// HyperbolicLoRA on gfx1250 (MI455X).
//   out = 0.25 * logmap_scale(row) * [ f0(row) * (x @ A^T) ] @ B^T
// Two rank-64 GEMMs via v_wmma_f32_16x16x32_f16; rowwise maps in f32.
// Weights are pre-packed into exact WMMA B-fragment order so that every
// fragment fetch is a fully coalesced, L2-resident stream.
// ---------------------------------------------------------------------------

typedef __attribute__((ext_vector_type(16))) _Float16 v16h;
typedef __attribute__((ext_vector_type(8)))  _Float16 v8h;
typedef __attribute__((ext_vector_type(8)))  float    v8f;

#define N_ROWS 16384
#define D_IN   4096
#define D_OUT  4096
#define RANK   64
#define EPSF   1e-7f

// -------- fragment builders (contiguous K order, same in A and B) -----------------
// A-matrix 16x32 f16: lanes 0-15 -> M=lane, K=[0,16);
//                     lanes 16-31 -> M=lane-16, K=[16,32).
__device__ __forceinline__ v16h load_a_frag(const _Float16* rowptr, int kbase, int half) {
  const v8h lo = *(const v8h*)(rowptr + kbase + half * 16);
  const v8h hi = *(const v8h*)(rowptr + kbase + half * 16 + 8);
  return __builtin_shufflevector(lo, hi, 0,1,2,3,4,5,6,7,8,9,10,11,12,13,14,15);
}
// B fragments are pre-packed: 16 contiguous halves per lane.
__device__ __forceinline__ v16h load_b_frag(const _Float16* p) {
  const v8h lo = *(const v8h*)(p);
  const v8h hi = *(const v8h*)(p + 8);
  return __builtin_shufflevector(lo, hi, 0,1,2,3,4,5,6,7,8,9,10,11,12,13,14,15);
}

// -------- kernel 0: convert + pack weights into WMMA fragment order ---------
// B-matrix 32x16 f16 layout: lane = N(mod 16); lanes 0-15 K=[0,16),
// lanes 16-31 K=[16,32) within each 32-wide K step.
// Apk: [kchunk 0..63][ni 0..3][st 0..1][lane 0..31][16 halves]   (from A_w)
// Bpk: [ntile  0..255]        [st 0..1][lane 0..31][16 halves]   (from B_w)
__global__ __launch_bounds__(256) void k_pack_weights(
    const float* __restrict__ A_w, const float* __restrict__ B_w,
    _Float16* __restrict__ Apk, _Float16* __restrict__ Bpk) {
  const int tid = blockIdx.x * 256 + threadIdx.x;   // 32768 threads
  const float* src;
  _Float16* dst;
  if (tid < 16384) {                                // A_w (64 x 4096)
    const int lane  = tid & 31;
    const int st    = (tid >> 5) & 1;
    const int ni    = (tid >> 6) & 3;
    const int chunk = tid >> 8;
    const int n = ni * 16 + (lane & 15);
    const int k = chunk * 64 + st * 32 + (lane >> 4) * 16;
    src = A_w + (size_t)n * D_IN + k;
    dst = Apk + (size_t)tid * 16;
  } else {                                          // B_w (4096 x 64)
    const int id   = tid - 16384;
    const int lane = id & 31;
    const int st   = (id >> 5) & 1;
    const int nt   = id >> 6;
    const int n = nt * 16 + (lane & 15);
    const int k = st * 32 + (lane >> 4) * 16;
    src = B_w + (size_t)n * RANK + k;
    dst = Bpk + (size_t)id * 16;
  }
  const float4 f0 = *(const float4*)(src);
  const float4 f1 = *(const float4*)(src + 4);
  const float4 f2 = *(const float4*)(src + 8);
  const float4 f3 = *(const float4*)(src + 12);
  v8h h0, h1;
  h0[0]=(_Float16)f0.x; h0[1]=(_Float16)f0.y; h0[2]=(_Float16)f0.z; h0[3]=(_Float16)f0.w;
  h0[4]=(_Float16)f1.x; h0[5]=(_Float16)f1.y; h0[6]=(_Float16)f1.z; h0[7]=(_Float16)f1.w;
  h1[0]=(_Float16)f2.x; h1[1]=(_Float16)f2.y; h1[2]=(_Float16)f2.z; h1[3]=(_Float16)f2.w;
  h1[4]=(_Float16)f3.x; h1[5]=(_Float16)f3.y; h1[6]=(_Float16)f3.z; h1[7]=(_Float16)f3.w;
  *(v8h*)(dst)     = h0;
  *(v8h*)(dst + 8) = h1;
}

// -------- kernel 1: s1 = f0(row) * (x @ A^T), stored f16 --------------------
// Grid: 512 WGs x 256 threads. WG = 32 rows x 64 cols, K loop over 4096.
// Double-buffered LDS stage: one barrier per 64-wide K chunk.
__global__ __launch_bounds__(256) void k_gemm1_expmap(
    const float* __restrict__ x, const _Float16* __restrict__ Apk,
    _Float16* __restrict__ s1h) {
  __shared__ _Float16 xs[2][32][72]; // double-buffered x chunk, f16, padded
  __shared__ float    rs[32];        // per-row sum of squares
  __shared__ float    f0s[32];       // per-row expmap scale

  const int t    = threadIdx.x;
  const int wave = t >> 5;
  const int lane = t & 31;
  const int l    = lane & 15;
  const int half = lane >> 4;
  const int mi   = wave >> 2;       // 0..1  (M tile)
  const int ni   = wave & 3;        // 0..3  (N tile)
  const int row0 = blockIdx.x * 32;

  const int lr = t >> 3;            // loader row 0..31
  const int lc = (t & 7) * 8;       // loader col base (8 floats per thread)

  if (t < 32) rs[t] = 0.0f;

  v8f c = {};
  float ss = 0.0f;
  const float* xrow = x + (size_t)(row0 + lr) * D_IN;

  // preload chunk 0
  float4 a0 = *(const float4*)(xrow + lc);
  float4 a1 = *(const float4*)(xrow + lc + 4);
  int buf = 0;

  for (int chunk = 0; chunk < 64; ++chunk) {
    // stage current chunk (f32 -> f16) into LDS; accumulate row sumsq
    ss += a0.x*a0.x + a0.y*a0.y + a0.z*a0.z + a0.w*a0.w
        + a1.x*a1.x + a1.y*a1.y + a1.z*a1.z + a1.w*a1.w;
    v8h h;
    h[0]=(_Float16)a0.x; h[1]=(_Float16)a0.y; h[2]=(_Float16)a0.z; h[3]=(_Float16)a0.w;
    h[4]=(_Float16)a1.x; h[5]=(_Float16)a1.y; h[6]=(_Float16)a1.z; h[7]=(_Float16)a1.w;
    *(v8h*)&xs[buf][lr][lc] = h;

    // issue next chunk's loads before the barrier (overlap with WMMA)
    if (chunk + 1 < 64) {
      const float* nxt = xrow + (chunk + 1) * 64 + lc;
      a0 = *(const float4*)(nxt);
      a1 = *(const float4*)(nxt + 4);
      if (chunk + 2 < 64) __builtin_prefetch(xrow + (chunk + 2) * 64 + lc, 0, 3);
    }
    __syncthreads();

    // WMMA on the staged chunk; weight fragments stream coalesced from L2
    const _Float16* apkc = Apk + ((size_t)(chunk * 4 + ni) * 2) * 512 + lane * 16;
    #pragma unroll
    for (int st = 0; st < 2; ++st) {
      const v16h af = load_a_frag(&xs[buf][mi * 16 + l][0], st * 32, half);
      const v16h bf = load_b_frag(apkc + st * 512);
      c = __builtin_amdgcn_wmma_f32_16x16x32_f16(false, af, false, bf,
                                                 (short)0, c, false, false);
    }
    buf ^= 1;
  }

  atomicAdd(&rs[lr], ss);
  __syncthreads();
  if (t < 32) {
    const float vn = fmaxf(sqrtf(rs[t]), EPSF);
    f0s[t] = sinhf(vn) / vn;              // expmap0 spatial scale
  }
  __syncthreads();

  // C layout: VGPR r, lanes 0-15 -> M=r ; lanes 16-31 -> M=8+r ; N=l
  #pragma unroll
  for (int r = 0; r < 8; ++r) {
    const int rin = mi * 16 + half * 8 + r;
    const float v = c[r] * f0s[rin];
    s1h[(size_t)(row0 + rin) * RANK + ni * 16 + l] = (_Float16)v;
  }
}

// -------- kernel 2: s2 = s1 @ B^T ; out = logmap_scale(row) * s2 ------------
// Grid: 512 WGs x 256 threads. WG = 32 rows x full N=4096, K=64.
// Pass 1 accumulates ||s2_row||^2, pass 2 recomputes tiles and writes output.
__global__ __launch_bounds__(256) void k_gemm2_logmap(
    const _Float16* __restrict__ s1h, const _Float16* __restrict__ Bpk,
    float* __restrict__ out) {
  __shared__ _Float16 s1s[32][72];
  __shared__ float    rq[32];       // per-row ||s2||^2
  __shared__ float    rsc[32];      // per-row output scale

  const int t    = threadIdx.x;
  const int wave = t >> 5;
  const int lane = t & 31;
  const int l    = lane & 15;
  const int half = lane >> 4;
  const int row0 = blockIdx.x * 32;

  // stage s1 tile (32x64 f16)
  {
    const int lr = t >> 3;
    const int lc = (t & 7) * 8;
    *(v8h*)&s1s[lr][lc] = *(const v8h*)(s1h + (size_t)(row0 + lr) * RANK + lc);
  }
  if (t < 32) rq[t] = 0.0f;
  __syncthreads();

  // A fragments: [M tile][K step], resident for the whole kernel
  v16h af[2][2];
  #pragma unroll
  for (int mt = 0; mt < 2; ++mt)
    #pragma unroll
    for (int st = 0; st < 2; ++st)
      af[mt][st] = load_a_frag(&s1s[mt * 16 + l][0], st * 32, half);

  // ---- pass 1: accumulate row sum-of-squares of s2 ----
  float qp[2][8];
  #pragma unroll
  for (int mt = 0; mt < 2; ++mt)
    #pragma unroll
    for (int r = 0; r < 8; ++r) qp[mt][r] = 0.0f;

  for (int it = 0; it < 32; ++it) {
    const int nt = it * 8 + wave;                 // N tile 0..255
    const _Float16* bp = Bpk + ((size_t)nt * 2) * 512 + lane * 16;
    const v16h b0 = load_b_frag(bp);
    const v16h b1 = load_b_frag(bp + 512);
    #pragma unroll
    for (int mt = 0; mt < 2; ++mt) {
      v8f c = {};
      c = __builtin_amdgcn_wmma_f32_16x16x32_f16(false, af[mt][0], false, b0,
                                                 (short)0, c, false, false);
      c = __builtin_amdgcn_wmma_f32_16x16x32_f16(false, af[mt][1], false, b1,
                                                 (short)0, c, false, false);
      #pragma unroll
      for (int r = 0; r < 8; ++r) qp[mt][r] += c[r] * c[r];
    }
  }
  // reduce over the 16 lanes of each half (rows stay separated by half)
  #pragma unroll
  for (int m = 1; m <= 8; m <<= 1)
    #pragma unroll
    for (int mt = 0; mt < 2; ++mt)
      #pragma unroll
      for (int r = 0; r < 8; ++r)
        qp[mt][r] += __shfl_xor(qp[mt][r], m, 32);
  if (l == 0) {
    #pragma unroll
    for (int mt = 0; mt < 2; ++mt)
      #pragma unroll
      for (int r = 0; r < 8; ++r)
        atomicAdd(&rq[mt * 16 + half * 8 + r], qp[mt][r]);
  }
  __syncthreads();
  if (t < 32) {
    const float q   = rq[t];
    const float yn  = fmaxf(sqrtf(q), EPSF);
    const float arg = fmaxf(sqrtf(1.0f + q), 1.0f + EPSF);
    rsc[t] = 0.25f * acoshf(arg) / yn;            // (ALPHA/RANK) * d / yn
  }
  __syncthreads();

  // ---- pass 2: recompute tiles (rank-64 GEMM is cheap), scale, store ----
  for (int it = 0; it < 32; ++it) {
    const int nt = it * 8 + wave;
    const int n  = nt * 16 + l;
    const _Float16* bp = Bpk + ((size_t)nt * 2) * 512 + lane * 16;
    const v16h b0 = load_b_frag(bp);
    const v16h b1 = load_b_frag(bp + 512);
    #pragma unroll
    for (int mt = 0; mt < 2; ++mt) {
      v8f c = {};
      c = __builtin_amdgcn_wmma_f32_16x16x32_f16(false, af[mt][0], false, b0,
                                                 (short)0, c, false, false);
      c = __builtin_amdgcn_wmma_f32_16x16x32_f16(false, af[mt][1], false, b1,
                                                 (short)0, c, false, false);
      #pragma unroll
      for (int r = 0; r < 8; ++r) {
        const int rin = mt * 16 + half * 8 + r;
        out[(size_t)(row0 + rin) * D_OUT + n] = c[r] * rsc[rin];
      }
    }
  }
}

// ---------------------------------------------------------------------------
extern "C" void kernel_launch(void* const* d_in, const int* in_sizes, int n_in,
                              void* d_out, int out_size, void* d_ws, size_t ws_size,
                              hipStream_t stream) {
  const float* x   = (const float*)d_in[0];   // (16384, 4096)
  const float* A_w = (const float*)d_in[1];   // (64, 4096)
  const float* B_w = (const float*)d_in[2];   // (4096, 64)
  float* out = (float*)d_out;                 // (16384, 4096)

  // workspace layout (f16): Apk | Bpk | s1h   -> 3 MB total
  _Float16* Apk = (_Float16*)d_ws;
  _Float16* Bpk = Apk + (size_t)RANK * D_IN;
  _Float16* s1h = Bpk + (size_t)D_OUT * RANK;

  k_pack_weights<<<dim3(128), dim3(256), 0, stream>>>(A_w, B_w, Apk, Bpk);
  k_gemm1_expmap<<<dim3(N_ROWS / 32), dim3(256), 0, stream>>>(x, Apk, s1h);
  k_gemm2_logmap<<<dim3(N_ROWS / 32), dim3(256), 0, stream>>>(s1h, Bpk, out);
}
